// Multi_EfficientAttention_43_56229711839613
// MI455X (gfx1250) — matrix operand order repaired and checked by reading, not compiled, not run on hardware
//
#include <hip/hip_runtime.h>
#include <hip/hip_bf16.h>
#include <stdint.h>

// Multi_EfficientAttention for MI455X (gfx1250, wave32, WMMA bf16)
//
// B=4, N=8192, C=256, H=8, DH=32.  All matmuls except ctx_kernel's via v_wmma_f32_16x16x32_bf16
// (fp32 accumulate).  Softmaxes in fp32.  Workspace layout (bytes):
//   [0,512K)      : WqT,WkT,WvT,WrT  bf16, stored K-major (N x K)
//   [512K,516K)   : kmax keys (monotonic-uint mapped fp32), B*C entries
//   [516K+pad ...): q_sm bf16 (16MB), k fp32 (32MB), v bf16 (16MB), ctxT bf16 (64KB)
// total ~64.7 MB of d_ws.

#define DIMC   256
#define HEADS  8
#define DHD    32
#define NTOK   8192
#define BATCH  4
#define BN     (BATCH*NTOK)     // 32768 tokens
#define MTILE  64               // tokens per block in GEMM kernels
#define KCH    32               // K chunk (one WMMA K)

typedef __attribute__((ext_vector_type(16))) __bf16 v16bf;
typedef __attribute__((ext_vector_type(8)))  __bf16 v8bf;
typedef __attribute__((ext_vector_type(8)))  float  v8f;

__device__ __forceinline__ unsigned short f2bf(float f) {
    unsigned u = __float_as_uint(f);
    unsigned r = u + 0x7FFFu + ((u >> 16) & 1u);   // round-to-nearest-even
    return (unsigned short)(r >> 16);
}
__device__ __forceinline__ float bf2f(unsigned short h) {
    return __uint_as_float(((unsigned)h) << 16);
}
// order-preserving float<->uint mapping so atomicMax(uint) == float max
__device__ __forceinline__ unsigned fkey(float x) {
    unsigned u = __float_as_uint(x);
    return (u & 0x80000000u) ? ~u : (u | 0x80000000u);
}
__device__ __forceinline__ float funkey(unsigned k) {
    return __uint_as_float((k & 0x80000000u) ? (k ^ 0x80000000u) : ~k);
}

__device__ __forceinline__ v16bf join16(v8bf lo, v8bf hi) {
    return __builtin_shufflevector(lo, hi, 0,1,2,3,4,5,6,7,8,9,10,11,12,13,14,15);
}

// A fragment (16x32 bf16, row-major src):
//   lane<16: row=lane,   K = {0..7, 16..23}
//   lane>=16: row=lane-16, K = {8..15, 24..31}
__device__ __forceinline__ v16bf load_a_frag(const unsigned short* base, int rowstride, int lane) {
    const int row  = lane & 15;
    const int koff = (lane >> 4) * 8;
    v8bf lo = *(const v8bf*)(base + row * rowstride + koff);
    v8bf hi = *(const v8bf*)(base + row * rowstride + 16 + koff);
    return join16(lo, hi);
}
// B fragment (32x16 bf16) from K-major storage Bt[n][k]:
//   lane<16: col=lane, K = {0..7, 16..23} ;  lane>=16: col=lane-16, K = {8..15, 24..31}
__device__ __forceinline__ v16bf load_b_frag(const unsigned short* base, int rowstride, int lane) {
    const int col = lane & 15;
    const int kb  = (lane >> 4) * 8;
    v8bf lo = *(const v8bf*)(base + col * rowstride + kb);
    v8bf hi = *(const v8bf*)(base + col * rowstride + kb + 16);
    return join16(lo, hi);
}

// ---------------------------------------------------------------------------
// Prep: transpose+convert 4 weight matrices (fp32 [K][N]) -> bf16 [N][K];
// init per-channel k-max keys to 0 (== -inf under fkey mapping).
// ---------------------------------------------------------------------------
__global__ __launch_bounds__(256) void prep_kernel(
    const float* __restrict__ Wq, const float* __restrict__ Wk,
    const float* __restrict__ Wv, const float* __restrict__ Wr,
    unsigned short* __restrict__ WTq, unsigned short* __restrict__ WTk,
    unsigned short* __restrict__ WTv, unsigned short* __restrict__ WTr,
    unsigned* __restrict__ kmax_keys)
{
    const int idx = blockIdx.x * blockDim.x + threadIdx.x;
    if (idx < 4 * DIMC * DIMC) {
        const int wi  = idx >> 16;
        const int rem = idx & 0xFFFF;
        const int n = rem >> 8, k = rem & 255;
        const float* W = (wi == 0) ? Wq : (wi == 1) ? Wk : (wi == 2) ? Wv : Wr;
        unsigned short* WT = (wi == 0) ? WTq : (wi == 1) ? WTk : (wi == 2) ? WTv : WTr;
        WT[n * DIMC + k] = f2bf(W[(size_t)k * DIMC + n]);
    } else if (idx < 4 * DIMC * DIMC + BATCH * DIMC) {
        kmax_keys[idx - 4 * DIMC * DIMC] = 0u;
    }
}

// ---------------------------------------------------------------------------
// Projection GEMM: out[M0..M0+64, 0..256) = (X (+Xpe)) @ WT^T + bias
// 256 threads = 8 waves; wave w owns output columns [w*32, w*32+32)
// (exactly one head for EPI==0), 4 M-subtiles of 16 rows each.
// EPI 0: q -> softmax over 32 head channels, write bf16
// EPI 1: k -> write fp32 + atomic per-(b,channel) max (mapped-uint)
// EPI 2: v -> write bf16
// ---------------------------------------------------------------------------
template <int EPI>
__global__ __launch_bounds__(256) void gemm_epi(
    const float* __restrict__ X, const float* __restrict__ Xpe,
    const unsigned short* __restrict__ WT, const float* __restrict__ bias,
    unsigned short* __restrict__ out_bf, float* __restrict__ out_f32,
    unsigned* __restrict__ kmax_keys)
{
    __shared__ unsigned short As[MTILE * KCH];   // 64x32 bf16 staging (4 KB)
    const int t    = threadIdx.x;
    const int lane = t & 31;
    const int w    = t >> 5;
    const int M0   = blockIdx.x * MTILE;
    const int col  = lane & 15;
    const int hi16 = lane >> 4;

    v8f acc[4][2];
#pragma unroll
    for (int mt = 0; mt < 4; ++mt)
#pragma unroll
        for (int ns = 0; ns < 2; ++ns)
            acc[mt][ns] = (v8f){0.f,0.f,0.f,0.f,0.f,0.f,0.f,0.f};

    const int lr = t >> 2;            // staging row 0..63
    const int lk = (t & 3) * 8;       // staging k  0,8,16,24

    for (int kc = 0; kc < DIMC / KCH; ++kc) {
        const int k0 = kc * KCH;
        __syncthreads();
        {
            const size_t gb = (size_t)(M0 + lr) * DIMC + k0 + lk;
            const float4 x0 = *(const float4*)(X + gb);
            const float4 x1 = *(const float4*)(X + gb + 4);
            float f[8] = {x0.x, x0.y, x0.z, x0.w, x1.x, x1.y, x1.z, x1.w};
            if (Xpe) {
                const float4 p0 = *(const float4*)(Xpe + gb);
                const float4 p1 = *(const float4*)(Xpe + gb + 4);
                f[0] += p0.x; f[1] += p0.y; f[2] += p0.z; f[3] += p0.w;
                f[4] += p1.x; f[5] += p1.y; f[6] += p1.z; f[7] += p1.w;
            }
            if (kc + 1 < DIMC / KCH) __builtin_prefetch(X + gb + KCH, 0, 1);
#pragma unroll
            for (int i = 0; i < 8; ++i) As[lr * KCH + lk + i] = f2bf(f[i]);
        }
        __syncthreads();

        v16bf bfr[2];
#pragma unroll
        for (int ns = 0; ns < 2; ++ns)
            bfr[ns] = load_b_frag(WT + (size_t)(w * 32 + ns * 16) * DIMC + k0, DIMC, lane);
#pragma unroll
        for (int mt = 0; mt < 4; ++mt) {
            v16bf afr = load_a_frag(&As[(mt * 16) * KCH], KCH, lane);
#pragma unroll
            for (int ns = 0; ns < 2; ++ns)
                acc[mt][ns] = __builtin_amdgcn_wmma_f32_16x16x32_bf16(
                    false, afr, false, bfr[ns], (short)0, acc[mt][ns], false, false);
        }
    }

    const float b0 = bias[w * 32 + col];
    const float b1 = bias[w * 32 + 16 + col];

    if (EPI == 0) {
        // softmax over this wave's 32 columns (== one head) per output row.
        // Row r's 32 values live in the 16-lane half-wave (hi16 group) at vgpr j.
#pragma unroll
        for (int mt = 0; mt < 4; ++mt) {
#pragma unroll
            for (int j = 0; j < 8; ++j) {
                float v0 = acc[mt][0][j] + b0;
                float v1 = acc[mt][1][j] + b1;
                float m = fmaxf(v0, v1);
#pragma unroll
                for (int d = 1; d < 16; d <<= 1) m = fmaxf(m, __shfl_xor(m, d, 32));
                float e0 = __expf(v0 - m), e1 = __expf(v1 - m);
                float s = e0 + e1;
#pragma unroll
                for (int d = 1; d < 16; d <<= 1) s += __shfl_xor(s, d, 32);
                const int tok = M0 + mt * 16 + j + hi16 * 8;
                out_bf[(size_t)tok * DIMC + w * 32 + col]      = f2bf(e0 / s);
                out_bf[(size_t)tok * DIMC + w * 32 + 16 + col] = f2bf(e1 / s);
            }
        }
    } else if (EPI == 1) {
        float cm0 = -3.4e38f, cm1 = -3.4e38f;
#pragma unroll
        for (int mt = 0; mt < 4; ++mt) {
#pragma unroll
            for (int j = 0; j < 8; ++j) {
                const float v0 = acc[mt][0][j] + b0;
                const float v1 = acc[mt][1][j] + b1;
                const int tok = M0 + mt * 16 + j + hi16 * 8;
                out_f32[(size_t)tok * DIMC + w * 32 + col]      = v0;
                out_f32[(size_t)tok * DIMC + w * 32 + 16 + col] = v1;
                cm0 = fmaxf(cm0, v0);
                cm1 = fmaxf(cm1, v1);
            }
        }
        cm0 = fmaxf(cm0, __shfl_xor(cm0, 16, 32));
        cm1 = fmaxf(cm1, __shfl_xor(cm1, 16, 32));
        const int bidx = M0 >> 13;   // 8192 tokens per batch
        if (hi16 == 0) atomicMax(&kmax_keys[bidx * DIMC + w * 32 + col], fkey(cm0));
        else           atomicMax(&kmax_keys[bidx * DIMC + w * 32 + 16 + col], fkey(cm1));
    } else {   // EPI == 2 : v, bf16
#pragma unroll
        for (int mt = 0; mt < 4; ++mt) {
#pragma unroll
            for (int j = 0; j < 8; ++j) {
                const int tok = M0 + mt * 16 + j + hi16 * 8;
                out_bf[(size_t)tok * DIMC + w * 32 + col]      = f2bf(acc[mt][0][j] + b0);
                out_bf[(size_t)tok * DIMC + w * 32 + 16 + col] = f2bf(acc[mt][1][j] + b1);
            }
        }
    }
}

// ---------------------------------------------------------------------------
// context[b,h,d,e] = sum_n exp(k[b,n,h,d]-max)*v[b,n,h,e] / sum_n exp(...)
// One block per (b,h).  Streams 64-token tiles through LDS.  Writes bf16
// transposed ctxT[bh][e][d] (K-major for the phase-1 WMMA of attn_out).
// ---------------------------------------------------------------------------
__global__ __launch_bounds__(256) void ctx_kernel(
    const float* __restrict__ kbuf, const unsigned short* __restrict__ vbuf,
    const unsigned* __restrict__ kmax_keys, unsigned short* __restrict__ ctxT)
{
    __shared__ float smax[DHD];
    __shared__ float sw[64][DHD];
    __shared__ float sv[64][DHD];
    __shared__ float sz[DHD];
    const int t  = threadIdx.x;
    const int bh = blockIdx.x;
    const int b = bh >> 3, h = bh & 7;
    const int c0 = h * DHD;
    if (t < DHD) smax[t] = funkey(kmax_keys[b * DIMC + c0 + t]);
    __syncthreads();

    const int d  = t & 31;      // context row (k-channel)
    const int eg = t >> 5;      // wave id -> 4-wide e group
    const int eb = eg * 4;
    float acc[4] = {0.f, 0.f, 0.f, 0.f};
    float z = 0.f;

    const int lr = t >> 2, lk = (t & 3) * 8;
    for (int nt = 0; nt < NTOK / 64; ++nt) {
        __syncthreads();
        {
            const size_t gb = ((size_t)b * NTOK + nt * 64 + lr) * DIMC + c0 + lk;
            const float4 k0 = *(const float4*)(kbuf + gb);
            const float4 k1 = *(const float4*)(kbuf + gb + 4);
            const float kv[8] = {k0.x, k0.y, k0.z, k0.w, k1.x, k1.y, k1.z, k1.w};
#pragma unroll
            for (int i = 0; i < 8; ++i) sw[lr][lk + i] = __expf(kv[i] - smax[lk + i]);
#pragma unroll
            for (int i = 0; i < 8; ++i) sv[lr][lk + i] = bf2f(vbuf[gb + i]);
        }
        __syncthreads();
#pragma unroll 4
        for (int n = 0; n < 64; ++n) {
            const float wv = sw[n][d];
            if (eg == 0) z += wv;
#pragma unroll
            for (int i = 0; i < 4; ++i) acc[i] += wv * sv[n][eb + i];
        }
    }
    if (eg == 0) sz[d] = z;
    __syncthreads();
#pragma unroll
    for (int i = 0; i < 4; ++i) {
        const int e = eb + i;
        ctxT[(size_t)bh * DHD * DHD + e * DHD + d] = f2bf(acc[i] / sz[d]);
    }
}

// ---------------------------------------------------------------------------
// Fused: att[m,h,e] = sum_d q_sm[m,h,d]*ctx[h,d,e]  (one WMMA per tile, K=32)
// then   out = att @ WrT^T + br                      (WMMA GEMM from LDS)
// ---------------------------------------------------------------------------
__global__ __launch_bounds__(256) void attn_out(
    const unsigned short* __restrict__ qsm, const unsigned short* __restrict__ ctxT,
    const unsigned short* __restrict__ WrT, const float* __restrict__ br,
    float* __restrict__ out)
{
    __shared__ unsigned short Att[MTILE * DIMC];   // 64x256 bf16 (32 KB)
    const int t    = threadIdx.x;
    const int lane = t & 31;
    const int w    = t >> 5;
    const int M0   = blockIdx.x * MTILE;
    const int b    = M0 >> 13;
    const int col  = lane & 15;
    const int hi16 = lane >> 4;

    // ---- phase 1: wave w handles head h = w ----
    {
        const int h = w;
        v16bf bfr[2];
#pragma unroll
        for (int ns = 0; ns < 2; ++ns)
            bfr[ns] = load_b_frag(ctxT + (size_t)(b * HEADS + h) * DHD * DHD
                                       + (size_t)(ns * 16) * DHD, DHD, lane);
#pragma unroll
        for (int mt = 0; mt < 4; ++mt) {
            const int row  = lane & 15;
            const int koff = hi16 * 8;
            const int tok  = M0 + mt * 16 + row;
            const v8bf lo = *(const v8bf*)(qsm + (size_t)tok * DIMC + h * DHD + koff);
            const v8bf hv = *(const v8bf*)(qsm + (size_t)tok * DIMC + h * DHD + 16 + koff);
            const v16bf afr = join16(lo, hv);
#pragma unroll
            for (int ns = 0; ns < 2; ++ns) {
                v8f c = (v8f){0.f,0.f,0.f,0.f,0.f,0.f,0.f,0.f};
                c = __builtin_amdgcn_wmma_f32_16x16x32_bf16(
                    false, afr, false, bfr[ns], (short)0, c, false, false);
#pragma unroll
                for (int j = 0; j < 8; ++j) {
                    const int r = mt * 16 + j + hi16 * 8;
                    Att[r * DIMC + h * DHD + ns * 16 + col] = f2bf(c[j]);
                }
            }
        }
    }
    __syncthreads();

    // ---- phase 2: reprojection GEMM from LDS ----
    v8f acc[4][2];
#pragma unroll
    for (int mt = 0; mt < 4; ++mt)
#pragma unroll
        for (int ns = 0; ns < 2; ++ns)
            acc[mt][ns] = (v8f){0.f,0.f,0.f,0.f,0.f,0.f,0.f,0.f};

    for (int kc = 0; kc < DIMC / KCH; ++kc) {
        const int k0 = kc * KCH;
        v16bf bfr[2];
#pragma unroll
        for (int ns = 0; ns < 2; ++ns)
            bfr[ns] = load_b_frag(WrT + (size_t)(w * 32 + ns * 16) * DIMC + k0, DIMC, lane);
#pragma unroll
        for (int mt = 0; mt < 4; ++mt) {
            v16bf afr = load_a_frag(&Att[(mt * 16) * DIMC + k0], DIMC, lane);
#pragma unroll
            for (int ns = 0; ns < 2; ++ns)
                acc[mt][ns] = __builtin_amdgcn_wmma_f32_16x16x32_bf16(
                    false, afr, false, bfr[ns], (short)0, acc[mt][ns], false, false);
        }
    }

    const float b0 = br[w * 32 + col];
    const float b1 = br[w * 32 + 16 + col];
#pragma unroll
    for (int mt = 0; mt < 4; ++mt) {
#pragma unroll
        for (int j = 0; j < 8; ++j) {
            const int tok = M0 + mt * 16 + j + hi16 * 8;
            out[(size_t)tok * DIMC + w * 32 + col]      = acc[mt][0][j] + b0;
            out[(size_t)tok * DIMC + w * 32 + 16 + col] = acc[mt][1][j] + b1;
        }
    }
}

// ---------------------------------------------------------------------------
extern "C" void kernel_launch(void* const* d_in, const int* in_sizes, int n_in,
                              void* d_out, int out_size, void* d_ws, size_t ws_size,
                              hipStream_t stream)
{
    const float* f3   = (const float*)d_in[0];
    const float* f4   = (const float*)d_in[1];
    const float* f3pe = (const float*)d_in[2];
    const float* f4pe = (const float*)d_in[3];
    const float* Wq = (const float*)d_in[4];  const float* bq = (const float*)d_in[5];
    const float* Wk = (const float*)d_in[6];  const float* bk = (const float*)d_in[7];
    const float* Wv = (const float*)d_in[8];  const float* bv = (const float*)d_in[9];
    const float* Wr = (const float*)d_in[10]; const float* br = (const float*)d_in[11];

    char* ws = (char*)d_ws;
    unsigned short* WTq  = (unsigned short*)(ws + 0);
    unsigned short* WTk  = (unsigned short*)(ws + 131072);
    unsigned short* WTv  = (unsigned short*)(ws + 262144);
    unsigned short* WTr  = (unsigned short*)(ws + 393216);
    unsigned*       kmax = (unsigned*)(ws + 524288);
    size_t off = 528384;                              // 16B-aligned
    unsigned short* qsm  = (unsigned short*)(ws + off); off += (size_t)BN * DIMC * 2;
    float*          kb_  = (float*)(ws + off);          off += (size_t)BN * DIMC * 4;
    unsigned short* vb_  = (unsigned short*)(ws + off); off += (size_t)BN * DIMC * 2;
    unsigned short* ctxT = (unsigned short*)(ws + off); // 64 KB

    prep_kernel<<<(4 * DIMC * DIMC + BATCH * DIMC + 255) / 256, 256, 0, stream>>>(
        Wq, Wk, Wv, Wr, WTq, WTk, WTv, WTr, kmax);
    gemm_epi<0><<<BN / MTILE, 256, 0, stream>>>(f3, f3pe, WTq, bq, qsm, nullptr, nullptr);
    gemm_epi<1><<<BN / MTILE, 256, 0, stream>>>(f4, f4pe, WTk, bk, nullptr, kb_, kmax);
    gemm_epi<2><<<BN / MTILE, 256, 0, stream>>>(f4, nullptr, WTv, bv, vb_, nullptr, nullptr);
    ctx_kernel<<<BATCH * HEADS, 256, 0, stream>>>(kb_, vb_, kmax, ctxT);
    attn_out<<<BN / MTILE, 256, 0, stream>>>(qsm, ctxT, WTr, br, (float*)d_out);
}
